// BahdanauAttention_37812892074666
// MI455X (gfx1250) — matrix operand order repaired and checked by reading, not compiled, not run on hardware
//
#include <hip/hip_runtime.h>
#include <hip/hip_bf16.h>
#include <math.h>

// Bahdanau attention, MI455X (gfx1250).
// keys@W2 (137 GFLOP) on v_wmma_f32_16x16x32_bf16. All f32->bf16 conversion
// hoisted out of the hot loop: W2 prepacked to fragment order in d_ws, keys
// staged into LDS in fragment order. tanh(q+k)@V fused in epilogue using the
// gfx1250 hardware v_tanh_f32 transcendental.

#define B_   32
#define L_   2048
#define ENC_ 1024
#define ATT_ 1024

typedef __attribute__((ext_vector_type(16))) __bf16        v16bf;
typedef __attribute__((ext_vector_type(8)))  __bf16        v8bf;
typedef __attribute__((ext_vector_type(8)))  unsigned short v8u;
typedef __attribute__((ext_vector_type(8)))  float          v8f;
typedef __attribute__((ext_vector_type(4)))  float          v4f;
typedef __attribute__((ext_vector_type(2)))  float          v2f;

__device__ __forceinline__ unsigned short f2bf(float f) {
  // f32 -> bf16 round-to-nearest-even (integer path, hoisted out of hot loops)
  unsigned u = __float_as_uint(f);
  u += 0x7FFFu + ((u >> 16) & 1u);
  return (unsigned short)(u >> 16);
}

__device__ __forceinline__ float fast_tanh(float x) {
#if __has_builtin(__builtin_amdgcn_tanhf)
  return __builtin_amdgcn_tanhf(x);      // gfx1250 v_tanh_f32 (TRANS)
#elif __has_builtin(__builtin_amdgcn_tanh_f32)
  return __builtin_amdgcn_tanh_f32(x);
#else
  return tanhf(x);
#endif
}

__device__ __forceinline__ v16bf cat16(v8u lo, v8u hi) {
  v8bf a = __builtin_bit_cast(v8bf, lo);
  v8bf b = __builtin_bit_cast(v8bf, hi);
  return __builtin_shufflevector(a, b, 0, 1, 2, 3, 4, 5, 6, 7,
                                       8, 9, 10, 11, 12, 13, 14, 15);
}

// ---------------------------------------------------------------------------
// Kernel 0: prepack W2 (f32, k-major) -> bf16 fragment order in d_ws.
// w2f[((nt*32 + kc)*32 + lane)*16 + j] = bf16(W2[kc*32 + (lane>>4)*16 + j][nt*16 + (lane&15)])
// One-time 4 MB read; makes every B-fragment two contiguous global_load_b128.
// ---------------------------------------------------------------------------
__global__ __launch_bounds__(256)
void w2pack_kernel(const float* __restrict__ W2, unsigned short* __restrict__ w2f) {
  const int idx  = blockIdx.x * 256 + threadIdx.x;   // 0 .. 1048575
  const int j    = idx & 15;
  const int lane = (idx >> 4) & 31;
  const int kc   = (idx >> 9) & 31;
  const int nt   = idx >> 14;
  const int n    = nt * 16 + (lane & 15);
  const int K    = kc * 32 + (lane >> 4) * 16 + j;
  w2f[idx] = f2bf(W2[(size_t)K * ATT_ + n]);
}

// ---------------------------------------------------------------------------
// Kernel 1: qproj = query @ W1 (32x1024) on WMMA. Tiny; conversions inline.
// ---------------------------------------------------------------------------
__global__ __launch_bounds__(32)
void qproj_kernel(const float* __restrict__ query, const float* __restrict__ W1,
                  float* __restrict__ qout) {
  const int nt   = blockIdx.x;        // 0..63
  const int mt   = blockIdx.y;        // 0..1
  const int lane = threadIdx.x;
  const int nl   = lane & 15;
  const int half = lane >> 4;
  const int row  = mt * 16 + nl;
  const int n    = nt * 16 + nl;

  v8f acc = {};
  for (int kc = 0; kc < ENC_ / 32; ++kc) {
    const int kb = kc * 32;
    v16bf af, bf;
#pragma unroll
    for (int j = 0; j < 16; ++j) {
      af[j] = __builtin_bit_cast(__bf16,
          f2bf(query[(size_t)row * ENC_ + kb + half * 16 + j]));
      bf[j] = __builtin_bit_cast(__bf16,
          f2bf(W1[(size_t)(kb + half * 16 + j) * ATT_ + n]));
    }
    acc = __builtin_amdgcn_wmma_f32_16x16x32_bf16(false, af, false, bf,
                                                  (short)0, acc, false, false);
  }
#pragma unroll
  for (int r = 0; r < 8; ++r)
    qout[(size_t)(mt * 16 + r + half * 8) * ATT_ + n] = acc[r];
}

// ---------------------------------------------------------------------------
// Kernel 2: fused scores. Block = 64 rows (one b, 64 consecutive l), 4 waves.
// keys tile staged in LDS in FRAGMENT order (128 KB bf16; CDNA5 LDS 320 KB).
// Each wave owns 16 N-tiles processed in pairs: A-fragment (2x ds_load_b128)
// reused for 2 WMMAs; B-fragments are 2x global_load_b128 from prepacked w2f.
// ---------------------------------------------------------------------------
#define MROWS 64

__global__ __launch_bounds__(128)
void scores_kernel(const float* __restrict__ keys,
                   const unsigned short* __restrict__ w2f,
                   const float* __restrict__ qproj, const float* __restrict__ V,
                   float* __restrict__ score) {
  // sAf[((s*32 + kc)*32 + lane)*16 + j] : A-fragment order, 4 subtiles
  __shared__ __align__(16) unsigned short sAf[4 * 32 * 32 * 16];  // 128 KB
  __shared__ float rowsum[MROWS];

  const int mblock = blockIdx.x;               // 0..1023
  const int b      = mblock >> 5;
  const int l0     = (mblock & 31) << 6;
  const int tid    = threadIdx.x;
  const int wave   = tid >> 5;
  const int lane   = tid & 31;
  const int nl     = lane & 15;
  const int half   = lane >> 4;

  if (tid < MROWS) rowsum[tid] = 0.0f;

  // Stage keys[b, l0..l0+63, :] -> bf16 fragment order, packed u32 writes.
  {
    const float* kbase = keys + ((size_t)b * L_ + l0) * ENC_;
    for (int i = tid; i < MROWS * (ENC_ / 2); i += 128) {
      const int m = i >> 9;               // row 0..63
      const int p = i & 511;              // k-pair 0..511
      const int k = p * 2;
      v2f kv = __builtin_nontemporal_load((const v2f*)(kbase + (size_t)m * ENC_) + p);
      const unsigned packed =
          (unsigned)f2bf(kv.x) | ((unsigned)f2bf(kv.y) << 16);
      const int kc = k >> 5, kk = k & 31;
      const int h  = kk >> 4;
      const int j  = kk & 15;                            // j even
      const int ln = h * 16 + (m & 15);
      const int s  = m >> 4;
      const int du = (((s * 32) + kc) * 32 + ln) * 16 + j;
      *(unsigned*)&sAf[du] = packed;
    }
  }
  __syncthreads();

  float rs[4][8];
#pragma unroll
  for (int s = 0; s < 4; ++s)
#pragma unroll
    for (int r = 0; r < 8; ++r) rs[s][r] = 0.0f;

  for (int p = 0; p < 8; ++p) {
    const int nt0 = p * 8 + wave;          // pair of N-tiles per wave
    const int nt1 = nt0 + 4;
    const int n0  = nt0 * 16 + nl;
    const int n1  = nt1 * 16 + nl;

    v8f acc0[4], acc1[4];
#pragma unroll
    for (int s = 0; s < 4; ++s) { acc0[s] = (v8f){}; acc1[s] = (v8f){}; }

    const v8u* pb0 =
        (const v8u*)(w2f + (((size_t)nt0 * 32) * 32 + lane) * 16);
    const v8u* pb1 =
        (const v8u*)(w2f + (((size_t)nt1 * 32) * 32 + lane) * 16);

    for (int kc = 0; kc < ENC_ / 32; ++kc) {
      const v16bf bf0 = cat16(pb0[kc * 64 + 0], pb0[kc * 64 + 1]);
      const v16bf bf1 = cat16(pb1[kc * 64 + 0], pb1[kc * 64 + 1]);
#pragma unroll
      for (int s = 0; s < 4; ++s) {
        const v8u* pa = (const v8u*)&sAf[(((s * 32) + kc) * 32 + lane) * 16];
        const v16bf af = cat16(pa[0], pa[1]);
        acc0[s] = __builtin_amdgcn_wmma_f32_16x16x32_bf16(
            false, af, false, bf0, (short)0, acc0[s], false, false);
        acc1[s] = __builtin_amdgcn_wmma_f32_16x16x32_bf16(
            false, af, false, bf1, (short)0, acc1[s], false, false);
      }
    }

    // Fused epilogue: tanh(q + k) * V[n] via hardware v_tanh_f32.
    const float qv0 = qproj[(size_t)b * ATT_ + n0], vv0 = V[n0];
    const float qv1 = qproj[(size_t)b * ATT_ + n1], vv1 = V[n1];
#pragma unroll
    for (int s = 0; s < 4; ++s)
#pragma unroll
      for (int r = 0; r < 8; ++r)
        rs[s][r] += fast_tanh(acc0[s][r] + qv0) * vv0 +
                    fast_tanh(acc1[s][r] + qv1) * vv1;
  }

  // Reduce over the 16 lanes of each half (each lane held distinct columns n)
#pragma unroll
  for (int s = 0; s < 4; ++s)
#pragma unroll
    for (int r = 0; r < 8; ++r) {
      float v = rs[s][r];
#pragma unroll
      for (int mask = 1; mask <= 8; mask <<= 1)
        v += __shfl_xor(v, mask, 32);
      rs[s][r] = v;
    }

  if (nl == 0) {                           // lanes 0 and 16 of each wave
#pragma unroll
    for (int s = 0; s < 4; ++s)
#pragma unroll
      for (int r = 0; r < 8; ++r)
        atomicAdd(&rowsum[s * 16 + r + half * 8], rs[s][r]);  // ds_add_f32
  }
  __syncthreads();

  if (tid < MROWS)
    score[(size_t)b * L_ + l0 + tid] = rowsum[tid];
}

// ---------------------------------------------------------------------------
// Kernel 3: softmax over L per batch; weights straight into d_out.
// ---------------------------------------------------------------------------
__global__ __launch_bounds__(256)
void softmax_kernel(const float* __restrict__ score, float* __restrict__ wout) {
  __shared__ float red[256];
  const int b = blockIdx.x, t = threadIdx.x;
  const float* s = score + (size_t)b * L_;

  float v[8], m = -3.402823466e38f;
#pragma unroll
  for (int i = 0; i < 8; ++i) { v[i] = s[t + i * 256]; m = fmaxf(m, v[i]); }
  red[t] = m; __syncthreads();
  for (int o = 128; o > 0; o >>= 1) {
    if (t < o) red[t] = fmaxf(red[t], red[t + o]);
    __syncthreads();
  }
  m = red[0]; __syncthreads();

  float e[8], sum = 0.0f;
#pragma unroll
  for (int i = 0; i < 8; ++i) { e[i] = __expf(v[i] - m); sum += e[i]; }
  red[t] = sum; __syncthreads();
  for (int o = 128; o > 0; o >>= 1) {
    if (t < o) red[t] += red[t + o];
    __syncthreads();
  }
  const float inv = 1.0f / red[0];
#pragma unroll
  for (int i = 0; i < 8; ++i) wout[(size_t)b * L_ + t + i * 256] = e[i] * inv;
}

// ---------------------------------------------------------------------------
// Kernel 4: partial context sums (L split 8 ways; deterministic, no atomics).
// values (256 MB) streamed once, NT hint, coalesced 128b loads.
// ---------------------------------------------------------------------------
__global__ __launch_bounds__(256)
void ctx_partial_kernel(const float* __restrict__ values,
                        const float* __restrict__ w,
                        float* __restrict__ partial) {
  __shared__ float sW[256];
  const int lc = blockIdx.x;    // 0..7
  const int b  = blockIdx.y;    // 0..31
  const int t  = threadIdx.x;

  sW[t] = w[(size_t)b * L_ + lc * 256 + t];
  __syncthreads();

  const v4f* vbase =
      (const v4f*)(values + ((size_t)b * L_ + (size_t)lc * 256) * ENC_);
  v4f acc = {};
  for (int i = 0; i < 256; ++i) {
    v4f val = __builtin_nontemporal_load(vbase + (size_t)i * (ENC_ / 4) + t);
    acc += sW[i] * val;
  }
  v4f* ptl = (v4f*)(partial + ((size_t)(b * 8 + lc)) * ENC_);
  ptl[t] = acc;
}

__global__ __launch_bounds__(256)
void ctx_reduce_kernel(const float* __restrict__ partial,
                       float* __restrict__ ctx) {
  const int idx = blockIdx.x * 256 + threadIdx.x;  // 0..32767
  const int b = idx >> 10, e = idx & 1023;
  float s = 0.0f;
#pragma unroll
  for (int c = 0; c < 8; ++c) s += partial[((size_t)(b * 8 + c) << 10) + e];
  ctx[idx] = s;
}

// ---------------------------------------------------------------------------
extern "C" void kernel_launch(void* const* d_in, const int* in_sizes, int n_in,
                              void* d_out, int out_size, void* d_ws, size_t ws_size,
                              hipStream_t stream) {
  const float* query  = (const float*)d_in[0];   // (32,1024)
  const float* keys   = (const float*)d_in[1];   // (32,2048,1024)
  const float* values = (const float*)d_in[2];   // (32,2048,1024)
  const float* W1     = (const float*)d_in[3];   // (1024,1024)
  const float* W2     = (const float*)d_in[4];   // (1024,1024)
  const float* V      = (const float*)d_in[5];   // (1024,1)

  float* out     = (float*)d_out;
  float* out_ctx = out;                 // (32,1024)   = 32768 floats
  float* out_w   = out + 32768;         // (32,2048,1) = 65536 floats

  float* ws       = (float*)d_ws;       // total use: ~3.4 MB
  float* ws_q     = ws;                 // 32768  : query @ W1
  float* ws_score = ws + 32768;         // 65536  : pre-softmax scores
  float* ws_part  = ws + 98304;         // 262144 : context partials (32x8x1024)
  unsigned short* ws_w2f = (unsigned short*)(ws + 360448);  // 2 MB packed W2

  w2pack_kernel<<<4096, 256, 0, stream>>>(W2, ws_w2f);
  qproj_kernel<<<dim3(64, 2), 32, 0, stream>>>(query, W1, ws_q);
  scores_kernel<<<1024, 128, 0, stream>>>(keys, ws_w2f, ws_q, V, ws_score);
  softmax_kernel<<<32, 256, 0, stream>>>(ws_score, out_w);
  ctx_partial_kernel<<<dim3(8, 32), 256, 0, stream>>>(values, out_w, ws_part);
  ctx_reduce_kernel<<<128, 256, 0, stream>>>(ws_part, out_ctx);
}
